// BahdanauAttention_47416438948233
// MI455X (gfx1250) — matrix operand order repaired and checked by reading, not compiled, not run on hardware
//
#include <hip/hip_runtime.h>

// Bahdanau attention, B=32, S=2048, H=1024, fp32 in/out.
// Fused bf16-WMMA GEMM+tanh+dot for scores (keys read once from HBM),
// softmax, then memory-bound weighted sum (keys read a second time).
// HBM floor ~2x256MB / 23.3 TB/s ~= 22us. MTILE=128 halves Ua L2 traffic
// (512 blocks x 2MB = 1GB) vs MTILE=64. tanh via v_tanh_f32 trans op.

#define Hdim 1024
#define Sdim 2048
#define Bdim 32
#define MTILE 128
#define LROW 1032  // bf16 elems per LDS row: H + 8 pad (2064 B stride, 16B aligned, bank-spread)

typedef __attribute__((ext_vector_type(16))) __bf16 v16bf;
typedef __attribute__((ext_vector_type(8)))  float v8f;

union Frag { uint4 q[2]; v16bf v; };

__device__ __forceinline__ unsigned bf16_rne(float f) {
  unsigned u = __float_as_uint(f);
  return (u + 0x7FFFu + ((u >> 16) & 1u)) >> 16;  // round-to-nearest-even
}
__device__ __forceinline__ unsigned pack_bf16x2(float a, float b) {
  return bf16_rne(a) | (bf16_rne(b) << 16);
}

// CDNA5 V_TANH_F32 transcendental (1 trans op vs ~25-op ocml expansion).
__device__ __forceinline__ float fast_tanh(float x) {
#if __has_builtin(__builtin_amdgcn_tanhf)
  return __builtin_amdgcn_tanhf(x);
#else
  float r;
  asm volatile("v_tanh_f32 %0, %1\n\ts_nop 1" : "=v"(r) : "v"(x));
  return r;
#endif
}

// ---------------- kernel 0: Ua_w fp32 -> bf16 (2MB out) ---------------------
__global__ __launch_bounds__(256) void cvt_bf16_kernel(const float* __restrict__ in,
                                                       unsigned* __restrict__ out, int n2) {
  int i = blockIdx.x * 256 + threadIdx.x;
  if (i < n2) {
    float2 v = ((const float2*)in)[i];
    out[i] = pack_bf16x2(v.x, v.y);
  }
}

// ---------------- kernel 1: qpb[b,o] = query[b,:]*Wa_w[o,:] + Wa_b[o] + Ua_b[o]
__global__ __launch_bounds__(256) void qproj_kernel(const float* __restrict__ query,
                                                    const float* __restrict__ Wa_w,
                                                    const float* __restrict__ Wa_b,
                                                    const float* __restrict__ Ua_b,
                                                    float* __restrict__ qpb) {
  __shared__ float q[Hdim];
  const int b = blockIdx.x;
  for (int i = threadIdx.x; i < Hdim; i += 256) q[i] = query[b * Hdim + i];
  __syncthreads();
  for (int o = threadIdx.x; o < Hdim; o += 256) {
    const float* w = Wa_w + (size_t)o * Hdim;
    float acc = 0.f;
    for (int k = 0; k < Hdim; k += 4) {
      float4 wv = *(const float4*)(w + k);
      acc += wv.x * q[k] + wv.y * q[k + 1] + wv.z * q[k + 2] + wv.w * q[k + 3];
    }
    qpb[b * Hdim + o] = acc + Wa_b[o] + Ua_b[o];
  }
}

// ---------------- kernel 2: fused scores = sum_n tanh(keys*Ua^T + qpb) * Va --
// Block: 128 rows of one batch, 8 waves. Keys tile cached in LDS as bf16
// (~264 KB of the 320 KB/workgroup LDS). Each wave owns a 16-wide N chunk per
// iteration and covers all 8 M-subtiles; K-loop of wmma 16x16x32 bf16.
__global__ __launch_bounds__(256) void scores_kernel(const float* __restrict__ keys,
                                                     const unsigned short* __restrict__ uab16,
                                                     const float* __restrict__ qpb,
                                                     const float* __restrict__ va,
                                                     float* __restrict__ scores) {
  __shared__ unsigned short lds_keys[MTILE * LROW];   // 264,192 B
  __shared__ float lds_score[8][MTILE];               // per-wave partials (no atomics)

  const int tid  = threadIdx.x;
  const int wave = tid >> 5;
  const int lane = tid & 31;
  const int nlo  = lane & 15;   // N within fragment / M within A fragment
  const int hi   = lane >> 4;   // lane half selects K sub-range per ISA layout
  const int row0 = blockIdx.x * MTILE;  // 128 | S so all rows share one batch
  const int b    = row0 / Sdim;

  for (int i = tid; i < 8 * MTILE; i += 256) (&lds_score[0][0])[i] = 0.0f;

  // Phase A: stream keys tile, convert fp32 -> bf16 into LDS (coalesced b128 loads)
  const float4* src = (const float4*)(keys + (size_t)row0 * Hdim);
  for (int i = tid; i < MTILE * (Hdim / 4); i += 256) {
    const int r = i >> 8, c4 = i & 255;
    float4 v = src[(size_t)r * (Hdim / 4) + c4];
    uint2 p = { pack_bf16x2(v.x, v.y), pack_bf16x2(v.z, v.w) };
    *(uint2*)&lds_keys[r * LROW + c4 * 4] = p;
  }
  __syncthreads();

  // Phase B: 8 N-iterations x (K=1024 in steps of 32) x 8 M-subtiles
  for (int ni = 0; ni < 8; ++ni) {
    const int n = ni * 128 + wave * 16 + nlo;
    v8f c[8] = {};
    const unsigned short* brow = uab16 + (size_t)n * Hdim;  // B = Ua_w[n][k], bf16
    for (int k0 = 0; k0 < Hdim; k0 += 32) {
      __builtin_prefetch(brow + k0 + 128, 0, 1);  // global_prefetch_b8 for next B chunk
      // B fragment (32x16 bf16): VGPR0-3: K=k0+8*hi.., VGPR4-7: +16 (matches A)
      Frag bf;
      const uint4* bp = (const uint4*)(brow + k0 + hi * 8);
      bf.q[0] = bp[0];
      bf.q[1] = bp[2];
#pragma unroll
      for (int t = 0; t < 8; ++t) {
        // A fragment (16x32 bf16): M = t*16 + nlo; VGPR0-3: K=k0+8*hi.., VGPR4-7: +16
        const unsigned short* arow = &lds_keys[(t * 16 + nlo) * LROW + k0 + hi * 8];
        Frag af;
        af.q[0] = *(const uint4*)(arow);        // ds_load_b128
        af.q[1] = *(const uint4*)(arow + 16);   // ds_load_b128
        c[t] = __builtin_amdgcn_wmma_f32_16x16x32_bf16(false, af.v, false, bf.v,
                                                       (short)0, c[t], false, false);
      }
    }
    // Epilogue: C layout => lane holds N=nlo, M = t*16 + e + 8*hi
    const float qv  = qpb[b * Hdim + n];
    const float vav = va[n];
#pragma unroll
    for (int t = 0; t < 8; ++t) {
#pragma unroll
      for (int e = 0; e < 8; ++e) {
        float val = fast_tanh(c[t][e] + qv) * vav;
        val += __shfl_xor(val, 1, 16);
        val += __shfl_xor(val, 2, 16);
        val += __shfl_xor(val, 4, 16);
        val += __shfl_xor(val, 8, 16);   // sum over the 16 N values in this fragment
        if (nlo == 0) lds_score[wave][t * 16 + e + hi * 8] += val;
      }
    }
  }
  __syncthreads();
  if (tid < MTILE) {
    float s = 0.f;
#pragma unroll
    for (int w = 0; w < 8; ++w) s += lds_score[w][tid];
    scores[row0 + tid] = s;  // Va_b omitted: softmax is shift-invariant
  }
}

// ---------------- kernel 3: softmax over S per batch, writes weights to d_out
__global__ __launch_bounds__(256) void softmax_kernel(const float* __restrict__ scores,
                                                      float* __restrict__ weights) {
  __shared__ float red[256];
  const int b = blockIdx.x, tid = threadIdx.x;
  const float* s = scores + b * Sdim;
  float m = -3.4e38f;
  for (int i = tid; i < Sdim; i += 256) m = fmaxf(m, s[i]);
  red[tid] = m;
  __syncthreads();
  for (int st = 128; st > 0; st >>= 1) {
    if (tid < st) red[tid] = fmaxf(red[tid], red[tid + st]);
    __syncthreads();
  }
  m = red[0];
  __syncthreads();
  float sum = 0.f;
  for (int i = tid; i < Sdim; i += 256) sum += __expf(s[i] - m);
  red[tid] = sum;
  __syncthreads();
  for (int st = 128; st > 0; st >>= 1) {
    if (tid < st) red[tid] += red[tid + st];
    __syncthreads();
  }
  const float inv = 1.f / red[0];
  for (int i = tid; i < Sdim; i += 256) weights[b * Sdim + i] = __expf(s[i] - m) * inv;
}

// ---------------- kernel Z: zero context (must re-zero every launch) --------
__global__ __launch_bounds__(256) void zero_kernel(float* __restrict__ p, int n) {
  int i = blockIdx.x * 256 + threadIdx.x;
  if (i < n) p[i] = 0.f;
}

// ---------------- kernel 4: context[b,h] = sum_s weights[b,s]*keys[b,s,h] ---
// Split-S for parallelism; register accumulation + one atomic add per output.
__global__ __launch_bounds__(256) void context_kernel(const float* __restrict__ keys,
                                                      const float* __restrict__ weights,
                                                      float* __restrict__ context) {
  const int b = blockIdx.x, sc = blockIdx.y;
  const int s0 = sc * 256;
  const int h4 = threadIdx.x;  // float4 column group, 256 * 4 = H
  float4 acc = {0.f, 0.f, 0.f, 0.f};
  const float4* kp = (const float4*)(keys + ((size_t)b * Sdim + s0) * Hdim);
  const float* wp = weights + b * Sdim + s0;
  for (int s = 0; s < 256; ++s) {
    const float w = wp[s];
    float4 kv = kp[(size_t)s * (Hdim / 4) + h4];
    acc.x += w * kv.x; acc.y += w * kv.y; acc.z += w * kv.z; acc.w += w * kv.w;
  }
  float* dst = context + b * Hdim + h4 * 4;
  atomicAdd(dst + 0, acc.x);
  atomicAdd(dst + 1, acc.y);
  atomicAdd(dst + 2, acc.z);
  atomicAdd(dst + 3, acc.w);
}

extern "C" void kernel_launch(void* const* d_in, const int* in_sizes, int n_in,
                              void* d_out, int out_size, void* d_ws, size_t ws_size,
                              hipStream_t stream) {
  (void)in_sizes; (void)n_in; (void)out_size; (void)ws_size;
  const float* query = (const float*)d_in[0];
  const float* keys  = (const float*)d_in[1];
  const float* Wa_w  = (const float*)d_in[2];
  const float* Wa_b  = (const float*)d_in[3];
  const float* Ua_w  = (const float*)d_in[4];
  const float* Ua_b  = (const float*)d_in[5];
  const float* Va_w  = (const float*)d_in[6];
  // d_in[7] (Va_b) intentionally unused: softmax is shift-invariant.

  // workspace layout (~2.4 MB total)
  unsigned* uab16 = (unsigned*)d_ws;                                        // 2 MB bf16 Ua
  float* qpb      = (float*)((char*)d_ws + (size_t)Hdim * Hdim * 2);        // 128 KB
  float* scores   = qpb + Bdim * Hdim;                                      // 256 KB

  float* context = (float*)d_out;           // [B, 1, H] flat
  float* weights = context + Bdim * Hdim;   // [B, 1, S] flat

  cvt_bf16_kernel<<<(Hdim * Hdim / 2 + 255) / 256, 256, 0, stream>>>(Ua_w, uab16,
                                                                     Hdim * Hdim / 2);
  qproj_kernel<<<Bdim, 256, 0, stream>>>(query, Wa_w, Wa_b, Ua_b, qpb);
  zero_kernel<<<(Bdim * Hdim + 255) / 256, 256, 0, stream>>>(context, Bdim * Hdim);
  scores_kernel<<<(Bdim * Sdim) / MTILE, 256, 0, stream>>>(
      keys, (const unsigned short*)uab16, qpb, Va_w, scores);
  softmax_kernel<<<Bdim, 256, 0, stream>>>(scores, weights);
  context_kernel<<<dim3(Bdim, Sdim / 256), 256, 0, stream>>>(keys, weights, context);
}
